// LSTM_42734924595878
// MI455X (gfx1250) — compile-verified, matrix operand order checked by reading, not run on hardware
//
#include <hip/hip_runtime.h>
#include <math.h>

// ---------------------------------------------------------------------------
// LSTM on MI455X (gfx1250): bf16 WMMA (16x16x32) for all matmuls.
//   - Weights pre-packed (once per call) into the WMMA B-operand lane layout.
//   - x pre-converted to bf16 if scratch allows (deterministic ws_size gate).
//   - K loops fully unrolled + software-pipelined (depth 1): the double
//     buffer rotates by SSA renaming (no v_mov chains), loads issue ahead of
//     the WMMAs that consume the previous fragments.
//   - One fused kernel per timestep: 4 gates (1 wave each) -> LDS exchange ->
//     c/h update. 512 stream-ordered launches give the recurrence ordering.
// ---------------------------------------------------------------------------

typedef __attribute__((ext_vector_type(16))) __bf16 v16bf;
typedef __attribute__((ext_vector_type(8)))  float  v8f;

struct FragU { unsigned int u[8]; };

#define BATCH 128
#define SEQ   512
#define INSZ  512
#define HID   1024
#define CAT   1536          // INSZ + HID
#define NKC   48            // CAT / 32  (K-blocks for gate GEMMs)
#define NKO   64            // 2*HID / 32 (K-blocks for output GEMM)
#define OUTSZ 512

__device__ __forceinline__ unsigned int f2bf_u(float f) {
  unsigned int u = __builtin_bit_cast(unsigned int, f);
  return (u + 0x7FFFu + ((u >> 16) & 1u)) >> 16;   // round-to-nearest-even
}
__device__ __forceinline__ unsigned short f2bf(float f) {
  return (unsigned short)f2bf_u(f);
}
__device__ __forceinline__ unsigned int pack2(float lo, float hi) {
  return f2bf_u(lo) | (f2bf_u(hi) << 16);
}

// A fragment (16x32 bf16, M x K) from an fp32 row-major row (fallback path).
// ISA layout: lane<16 holds K in {k0..k0+7} u {k0+16..k0+23}; lane>=16 is +8.
__device__ __forceinline__ v16bf load_a_f32(const float* __restrict__ row,
                                            int k0, int lane) {
  const int koff = (lane & 16) ? 8 : 0;
  const float4* p = (const float4*)(row + k0 + koff);
  float4 a0 = p[0], a1 = p[1];     // run0: 8 floats
  float4 b0 = p[4], b1 = p[5];     // run1: +16 elements
  FragU f;
  f.u[0] = pack2(a0.x, a0.y); f.u[1] = pack2(a0.z, a0.w);
  f.u[2] = pack2(a1.x, a1.y); f.u[3] = pack2(a1.z, a1.w);
  f.u[4] = pack2(b0.x, b0.y); f.u[5] = pack2(b0.z, b0.w);
  f.u[6] = pack2(b1.x, b1.y); f.u[7] = pack2(b1.z, b1.w);
  return __builtin_bit_cast(v16bf, f);
}

// A fragment from a bf16 (ushort) row-major row: 2x contiguous uint4.
__device__ __forceinline__ v16bf load_a_bf16(const unsigned short* __restrict__ row,
                                             int k0, int lane) {
  const int koff = (lane & 16) ? 8 : 0;
  const uint4* p = (const uint4*)(row + k0 + koff); // 8 bf16 = 16B
  uint4 r0 = p[0], r1 = p[2];                       // run1 at +16 elems = +2 uint4
  FragU f;
  f.u[0] = r0.x; f.u[1] = r0.y; f.u[2] = r0.z; f.u[3] = r0.w;
  f.u[4] = r1.x; f.u[5] = r1.y; f.u[6] = r1.z; f.u[7] = r1.w;
  return __builtin_bit_cast(v16bf, f);
}

// B fragment from pre-packed buffer: frag (nblk,kblk) is 32 lanes x 32 bytes.
__device__ __forceinline__ v16bf load_b_frag(const unsigned short* __restrict__ pg,
                                             int nblk, int kblk, int nk, int lane) {
  const uint4* p = (const uint4*)(pg + (((size_t)nblk * nk + kblk) * 32 + lane) * 16);
  uint4 r0 = p[0], r1 = p[1];
  FragU f;
  f.u[0] = r0.x; f.u[1] = r0.y; f.u[2] = r0.z; f.u[3] = r0.w;
  f.u[4] = r1.x; f.u[5] = r1.y; f.u[6] = r1.z; f.u[7] = r1.w;
  return __builtin_bit_cast(v16bf, f);
}

#define WMMA_BF16(A, B, C) \
  __builtin_amdgcn_wmma_f32_16x16x32_bf16(false, (A), false, (B), (short)0, (C), false, false)

// ---------------------------------------------------------------------------
// Pack fp32 weight matrix W[K][N] (row stride ldn) into bf16 WMMA-B fragments.
// ---------------------------------------------------------------------------
__global__ void pack_b_kernel(const float* __restrict__ W,
                              unsigned short* __restrict__ out,
                              int nkb, int total, int ldn) {
  for (int idx = blockIdx.x * blockDim.x + threadIdx.x; idx < total;
       idx += gridDim.x * blockDim.x) {
    int e    = idx & 15;
    int lane = (idx >> 4) & 31;
    int tmp  = idx >> 9;
    int kblk = tmp % nkb;
    int nblk = tmp / nkb;
    int koff = (lane & 16) ? 8 : 0;
    int k = kblk * 32 + koff + ((e < 8) ? e : (8 + e));
    int n = nblk * 16 + (lane & 15);
    out[idx] = f2bf(W[(size_t)k * ldn + n]);
  }
}

// fp32 -> bf16 bulk convert (for x), grid-stride.
__global__ void cvt_bf16_kernel(const float* __restrict__ src,
                                unsigned short* __restrict__ dst, int total) {
  for (int idx = blockIdx.x * blockDim.x + threadIdx.x; idx < total;
       idx += gridDim.x * blockDim.x)
    dst[idx] = f2bf(src[idx]);
}

__global__ void init_state(unsigned short* __restrict__ hb0,
                           float* __restrict__ c0) {
  int idx = blockIdx.x * blockDim.x + threadIdx.x;
  if (idx < BATCH * HID) { hb0[idx] = 0; c0[idx] = 0.0f; }
}

// ---------------------------------------------------------------------------
// One LSTM timestep, fully fused. Grid (8, 32) x 128 threads.
//   blockIdx.x: batch tile (16 rows), blockIdx.y: 32-column block of H.
//   wave g (0..3) computes gate g's 16x32 tile (2 accs, K=1536 in 48 steps),
//   fully unrolled with depth-1 software pipeline.
// ---------------------------------------------------------------------------
__global__ __launch_bounds__(128) void lstm_step_kernel(
    const float* __restrict__ x,            // [B, T, INSZ] fp32
    const unsigned short* __restrict__ xb,  // [B, T, INSZ] bf16 (may be unused)
    const unsigned short* __restrict__ pW,  // [4][64nb][48kb][32][16] bf16
    const float* __restrict__ bi, const float* __restrict__ bff,
    const float* __restrict__ bo, const float* __restrict__ bc,
    const unsigned short* __restrict__ hb_cur,  // [B, H] bf16
    const float* __restrict__ c_cur,            // [B, H] fp32
    unsigned short* __restrict__ hb_nxt,
    float* __restrict__ c_nxt,
    float* __restrict__ hf_out,                 // [B, H] fp32 (latest h)
    int t, int use_xb) {
  __shared__ float lds[4][16][32];   // activated gate tiles, 8 KB

  const int tid  = threadIdx.x;
  const int lane = tid & 31;
  const int wave = tid >> 5;                 // gate id: 0=i 1=f 2=o 3=c~
  const int bm   = blockIdx.x;               // batch tile
  const int nb   = blockIdx.y;               // 32-col block
  const int mrow = bm * 16 + (lane & 15);    // batch row for A fragment

  const float*          xrow  = x  + ((size_t)mrow * SEQ + t) * INSZ;
  const unsigned short* xbrow = xb + ((size_t)mrow * SEQ + t) * INSZ;
  const unsigned short* hrow  = hb_cur + (size_t)mrow * HID;
  const unsigned short* wg    = pW + (size_t)wave * ((size_t)CAT * HID);
  const int nblk0 = nb * 2;

  v8f acc0 = {}, acc1 = {};
  v16bf a, b0, b1;

  // One pipeline stage: consume (a,b0,b1), prefetch next fragments.
#define STEP(NEXT_A, KB1)                                   \
  {                                                          \
    v16bf an  = (NEXT_A);                                    \
    v16bf bn0 = load_b_frag(wg, nblk0 + 0, (KB1), NKC, lane);\
    v16bf bn1 = load_b_frag(wg, nblk0 + 1, (KB1), NKC, lane);\
    acc0 = WMMA_BF16(a, b0, acc0);                           \
    acc1 = WMMA_BF16(a, b1, acc1);                           \
    a = an; b0 = bn0; b1 = bn1;                              \
  }

  // ---- x phase: kb = 0..15 (uniform branch hoisted out of the loop) ----
  b0 = load_b_frag(wg, nblk0 + 0, 0, NKC, lane);
  b1 = load_b_frag(wg, nblk0 + 1, 0, NKC, lane);
  if (use_xb) {
    a = load_a_bf16(xbrow, 0, lane);
#pragma unroll
    for (int kb = 0; kb < 15; ++kb)
      STEP(load_a_bf16(xbrow, (kb + 1) * 32, lane), kb + 1)
  } else {
    a = load_a_f32(xrow, 0, lane);
#pragma unroll
    for (int kb = 0; kb < 15; ++kb)
      STEP(load_a_f32(xrow, (kb + 1) * 32, lane), kb + 1)
  }
  // bridge: consume kb=15, prefetch first h fragment (kb=16)
  STEP(load_a_bf16(hrow, 0, lane), 16)

  // ---- h phase: kb = 16..47 ----
#pragma unroll
  for (int kb = 16; kb < NKC - 1; ++kb)
    STEP(load_a_bf16(hrow, (kb + 1) * 32 - INSZ, lane), kb + 1)
  acc0 = WMMA_BF16(a, b0, acc0);
  acc1 = WMMA_BF16(a, b1, acc1);
#undef STEP

  // Bias + activation, stash gate tile in LDS.
  // D layout: VGPR r -> M = r + (lane>=16 ? 8 : 0), N = (lane&15).
  const float* bias = (wave == 0) ? bi : (wave == 1) ? bff : (wave == 2) ? bo : bc;
  const int ncol = lane & 15;
  const int mr   = (lane >> 4) << 3;
#pragma unroll
  for (int nt = 0; nt < 2; ++nt) {
    float bv = bias[nb * 32 + nt * 16 + ncol];
#pragma unroll
    for (int r = 0; r < 8; ++r) {
      float v = (nt == 0 ? acc0[r] : acc1[r]) + bv;
      if (wave < 3) v = 1.0f / (1.0f + __expf(-v));  // sigmoid: i, f, o
      else          v = tanhf(v);                    // c~
      lds[wave][mr + r][nt * 16 + ncol] = v;
    }
  }
  __syncthreads();

  // Elementwise state update for this block's 16x32 patch.
  for (int e = tid; e < 16 * 32; e += 128) {
    int m = e >> 5, n = e & 31;
    float iv = lds[0][m][n], fv = lds[1][m][n];
    float ov = lds[2][m][n], cv = lds[3][m][n];
    size_t idx = (size_t)(bm * 16 + m) * HID + (nb * 32 + n);
    float cnew = fv * c_cur[idx] + iv * cv;
    float hnew = ov * tanhf(cnew);
    c_nxt[idx]  = cnew;
    hf_out[idx] = hnew;
    hb_nxt[idx] = f2bf(hnew);
  }
}

// ---------------------------------------------------------------------------
// output = [h, c] @ W_out + b_out.  Grid (8, 4) x 128; wave -> 2 n-tiles.
// Fully unrolled, pipelined like the step kernel.
// ---------------------------------------------------------------------------
__global__ __launch_bounds__(128) void lstm_out_kernel(
    const float* __restrict__ hf, const float* __restrict__ cf,
    const unsigned short* __restrict__ pWo, const float* __restrict__ bout,
    float* __restrict__ out) {
  const int tid  = threadIdx.x;
  const int lane = tid & 31;
  const int wave = tid >> 5;
  const int bm   = blockIdx.x;
  const int nblk0 = (blockIdx.y * 4 + wave) * 2;   // 0,2,..,30
  const int mrow = bm * 16 + (lane & 15);

  const float* hrow = hf + (size_t)mrow * HID;
  const float* crow = cf + (size_t)mrow * HID;

  v8f acc0 = {}, acc1 = {};
  v16bf a, b0, b1;

#define OSTEP(NEXT_A, KB1)                                    \
  {                                                            \
    v16bf an  = (NEXT_A);                                      \
    v16bf bn0 = load_b_frag(pWo, nblk0 + 0, (KB1), NKO, lane); \
    v16bf bn1 = load_b_frag(pWo, nblk0 + 1, (KB1), NKO, lane); \
    acc0 = WMMA_BF16(a, b0, acc0);                             \
    acc1 = WMMA_BF16(a, b1, acc1);                             \
    a = an; b0 = bn0; b1 = bn1;                                \
  }

  a  = load_a_f32(hrow, 0, lane);
  b0 = load_b_frag(pWo, nblk0 + 0, 0, NKO, lane);
  b1 = load_b_frag(pWo, nblk0 + 1, 0, NKO, lane);
#pragma unroll
  for (int kb = 0; kb < 31; ++kb)
    OSTEP(load_a_f32(hrow, (kb + 1) * 32, lane), kb + 1)
  OSTEP(load_a_f32(crow, 0, lane), 32)        // bridge into c half
#pragma unroll
  for (int kb = 32; kb < NKO - 1; ++kb)
    OSTEP(load_a_f32(crow, (kb + 1 - 32) * 32, lane), kb + 1)
  acc0 = WMMA_BF16(a, b0, acc0);
  acc1 = WMMA_BF16(a, b1, acc1);
#undef OSTEP

  const int ncol = lane & 15;
  const int mr   = (lane >> 4) << 3;
#pragma unroll
  for (int nt = 0; nt < 2; ++nt) {
    float bv = bout[(nblk0 + nt) * 16 + ncol];
#pragma unroll
    for (int r = 0; r < 8; ++r)
      out[(size_t)(bm * 16 + mr + r) * OUTSZ + (nblk0 + nt) * 16 + ncol] =
          (nt == 0 ? acc0[r] : acc1[r]) + bv;
  }
}

__global__ void copy_h_kernel(const float* __restrict__ hf, float* __restrict__ dst) {
  int idx = blockIdx.x * blockDim.x + threadIdx.x;
  if (idx < BATCH * HID) dst[idx] = hf[idx];
}

// ---------------------------------------------------------------------------
extern "C" void kernel_launch(void* const* d_in, const int* in_sizes, int n_in,
                              void* d_out, int out_size, void* d_ws, size_t ws_size,
                              hipStream_t stream) {
  (void)in_sizes; (void)n_in; (void)out_size;
  const float* x     = (const float*)d_in[0];
  const float* W_i   = (const float*)d_in[1];
  const float* b_i   = (const float*)d_in[2];
  const float* W_f   = (const float*)d_in[3];
  const float* b_f   = (const float*)d_in[4];
  const float* W_o   = (const float*)d_in[5];
  const float* b_o   = (const float*)d_in[6];
  const float* W_c   = (const float*)d_in[7];
  const float* b_c   = (const float*)d_in[8];
  const float* W_out = (const float*)d_in[9];
  const float* b_out = (const float*)d_in[10];

  // ---- scratch carve (all offsets well-aligned) ----
  char* ws = (char*)d_ws;
  size_t off = 0;
  unsigned short* pW  = (unsigned short*)(ws + off); off += (size_t)4 * CAT * HID * 2;    // 12.6 MB
  unsigned short* pWo = (unsigned short*)(ws + off); off += (size_t)2 * HID * OUTSZ * 2;  // 2 MB
  unsigned short* hb0 = (unsigned short*)(ws + off); off += (size_t)BATCH * HID * 2;
  unsigned short* hb1 = (unsigned short*)(ws + off); off += (size_t)BATCH * HID * 2;
  float* c0 = (float*)(ws + off); off += (size_t)BATCH * HID * 4;
  float* c1 = (float*)(ws + off); off += (size_t)BATCH * HID * 4;
  float* hf = (float*)(ws + off); off += (size_t)BATCH * HID * 4;   // 16 MiB so far
  unsigned short* xb = (unsigned short*)(ws + off);
  const size_t xb_bytes = (size_t)BATCH * SEQ * INSZ * 2;           // 67 MB
  const int use_xb = (ws_size >= off + xb_bytes) ? 1 : 0;           // deterministic

  // ---- pack weights to bf16 WMMA-B layout ----
  const size_t gstride = (size_t)CAT * HID;     // packed elems per gate
  const int gateTotal  = CAT * HID;
  pack_b_kernel<<<4096, 256, 0, stream>>>(W_i, pW + 0 * gstride, NKC, gateTotal, HID);
  pack_b_kernel<<<4096, 256, 0, stream>>>(W_f, pW + 1 * gstride, NKC, gateTotal, HID);
  pack_b_kernel<<<4096, 256, 0, stream>>>(W_o, pW + 2 * gstride, NKC, gateTotal, HID);
  pack_b_kernel<<<4096, 256, 0, stream>>>(W_c, pW + 3 * gstride, NKC, gateTotal, HID);
  pack_b_kernel<<<4096, 256, 0, stream>>>(W_out, pWo, NKO, 2 * HID * OUTSZ, OUTSZ);

  if (use_xb)
    cvt_bf16_kernel<<<8192, 256, 0, stream>>>(x, xb, BATCH * SEQ * INSZ);

  init_state<<<(BATCH * HID + 255) / 256, 256, 0, stream>>>(hb0, c0);

  // ---- 512 fused timesteps, ping-pong state buffers ----
  for (int t = 0; t < SEQ; ++t) {
    const unsigned short* hbc = (t & 1) ? hb1 : hb0;
    unsigned short*       hbn = (t & 1) ? hb0 : hb1;
    const float*          cc  = (t & 1) ? c1 : c0;
    float*                cn  = (t & 1) ? c0 : c1;
    lstm_step_kernel<<<dim3(8, 32), 128, 0, stream>>>(
        x, xb, pW, b_i, b_f, b_o, b_c, hbc, cc, hbn, cn, hf, t, use_xb);
  }
  // after t=511, final c is in c0, final h in hf.

  lstm_out_kernel<<<dim3(8, 4), 128, 0, stream>>>(hf, c0, pWo, b_out, (float*)d_out);
  copy_h_kernel<<<(BATCH * HID + 255) / 256, 256, 0, stream>>>(
      hf, (float*)d_out + BATCH * OUTSZ);
}
